// SelfAttentionHead_2027224564290
// MI455X (gfx1250) — matrix operand order repaired and checked by reading, not compiled, not run on hardware
//
#include <hip/hip_runtime.h>

// ---------------------------------------------------------------------------
// Single-head causal self-attention for MI455X (gfx1250, wave32, WMMA).
//   B=4, T=4096, C=1024, H=128.  All matrix math via v_wmma_f32_16x16x32_f16.
//   Flash-attention style; K/V chunks staged in LDS via gfx1250 async loads
//   (ASYNCcnt) shared by the 8 waves of each workgroup.
// ---------------------------------------------------------------------------

typedef __attribute__((ext_vector_type(16))) _Float16 v16h;
typedef __attribute__((ext_vector_type(8)))  float    v8f;
typedef int v4i_ __attribute__((vector_size(16)));   // matches builtin's pointee

#define B_ 4
#define T_ 4096
#define C_ 1024
#define H_ 128

union V16H { v16h v; _Float16 h[16]; uint4 u[2]; };
union H8   { uint4 u; _Float16 h[8]; };
union H4   { uint2 u; _Float16 h[4]; };

// ---- gfx1250 async global->LDS copy (ASYNCcnt), with safe fallback --------
#if __has_builtin(__builtin_amdgcn_global_load_async_to_lds_b128) && \
    __has_builtin(__builtin_amdgcn_s_wait_asynccnt)
#define USE_ASYNC_LDS 1
#else
#define USE_ASYNC_LDS 0
#endif

__device__ __forceinline__ void cp16_to_lds(void* lds, const _Float16* g) {
#if USE_ASYNC_LDS
  // Flat LDS pointers map to LDS by truncation to addr[31:0] (ISA 10.2), so
  // the low 32 bits of the generic pointer are the as(3) offset.
  __builtin_amdgcn_global_load_async_to_lds_b128(
      (__attribute__((address_space(1))) v4i_*)(uintptr_t)g,
      (__attribute__((address_space(3))) v4i_*)(unsigned int)(uintptr_t)lds,
      0, 0);
#else
  *(uint4*)lds = *(const uint4*)g;   // sync fallback: global_load + ds_store
#endif
}

__device__ __forceinline__ void wait_lds_cp() {
#if USE_ASYNC_LDS
  __builtin_amdgcn_s_wait_asynccnt(0);
#endif
}

// ---------------------------------------------------------------------------
// Kernel 1: convert x (f32) -> xh (f16), vectorized 4-wide.
// ---------------------------------------------------------------------------
__global__ __launch_bounds__(256) void cvt_x_kernel(const float* __restrict__ x,
                                                    _Float16* __restrict__ xh,
                                                    int n4) {
  int i = blockIdx.x * blockDim.x + threadIdx.x;
  const int stride = gridDim.x * blockDim.x;
  const float4* x4 = (const float4*)x;
  uint2* out2 = (uint2*)xh;
  for (; i < n4; i += stride) {
    float4 v = x4[i];
    H4 t;
    t.h[0] = (_Float16)v.x; t.h[1] = (_Float16)v.y;
    t.h[2] = (_Float16)v.z; t.h[3] = (_Float16)v.w;
    out2[i] = t.u;
  }
}

// ---------------------------------------------------------------------------
// Kernel 2: transpose+convert the three weight matrices.
//   wt layout: [3][H][C] f16, z: 0=key, 1=query, 2=value.  wt[z][n][k] = W[k][n]
// ---------------------------------------------------------------------------
__global__ __launch_bounds__(256) void prep_w_kernel(const float* __restrict__ wk,
                                                     const float* __restrict__ wq,
                                                     const float* __restrict__ wv,
                                                     _Float16* __restrict__ wt,
                                                     int total) {
  int i = blockIdx.x * blockDim.x + threadIdx.x;
  const int stride = gridDim.x * blockDim.x;
  for (; i < total; i += stride) {
    int z   = i / (H_ * C_);
    int rem = i - z * (H_ * C_);
    int n   = rem / C_;
    int k   = rem - n * C_;
    const float* w = (z == 0) ? wk : ((z == 1) ? wq : wv);
    wt[i] = (_Float16)w[k * H_ + n];
  }
}

// ---------------------------------------------------------------------------
// Kernel 3: QKV projection GEMM.  One 16x16 output tile per wave.
//   A layout (16x32 f16): lane(0-15)=row M; halves 0-7 -> K=8g..8g+7,
//   halves 8-15 -> K=8g+16..8g+23 (g = lane>>4).  B: lane&15 = col N,
//   same K mapping as A.  C/D: lane&15 = N, reg r -> M = r + 8g.
//   Outputs: z0 -> K [B*T,H] f16; z1 -> Q [B*T,H] f16; z2 -> V^T [B,H,T] f16.
// ---------------------------------------------------------------------------
__global__ __launch_bounds__(256) void qkv_gemm_kernel(const _Float16* __restrict__ xh,
                                                       const _Float16* __restrict__ wt,
                                                       _Float16* __restrict__ kh,
                                                       _Float16* __restrict__ qh,
                                                       _Float16* __restrict__ vt) {
  const int wave = threadIdx.x >> 5;
  const int lane = threadIdx.x & 31;
  const int g    = lane >> 4;
  const int l15  = lane & 15;
  const int mtile = blockIdx.x * 8 + wave;   // 0..1023
  const int ntile = blockIdx.y;              // 0..7
  const int z     = blockIdx.z;              // 0..2

  const _Float16* w    = wt + (size_t)z * (H_ * C_);
  const _Float16* arow = xh + (size_t)(mtile * 16 + l15) * C_;
  const _Float16* brow = w  + (size_t)(ntile * 16 + l15) * C_;

  v8f acc = {};
  for (int kc = 0; kc < C_; kc += 32) {
    V16H a, b;
    a.u[0] = *(const uint4*)(arow + kc + 8 * g);
    a.u[1] = *(const uint4*)(arow + kc + 8 * g + 16);
    b.u[0] = *(const uint4*)(brow + kc + 8 * g);
    b.u[1] = *(const uint4*)(brow + kc + 8 * g + 16);
    acc = __builtin_amdgcn_wmma_f32_16x16x32_f16(false, a.v, false, b.v,
                                                 (short)0, acc, false, false);
  }

  if (z == 2) {
    // V stored transposed: vt[b][h][t]; per lane the 8 acc rows are 8
    // consecutive tokens -> one packed 16B store.
    const int b    = mtile >> 8;                       // mtile*16 / T_
    const int t0   = mtile * 16 - b * T_ + 8 * g;
    const int hcol = ntile * 16 + l15;
    H8 p;
    #pragma unroll
    for (int r = 0; r < 8; ++r) p.h[r] = (_Float16)acc[r];
    *(uint4*)(vt + (size_t)(b * H_ + hcol) * T_ + t0) = p.u;
  } else {
    _Float16* dst = (z == 0) ? kh : qh;
    #pragma unroll
    for (int r = 0; r < 8; ++r)
      dst[(size_t)(mtile * 16 + r + 8 * g) * H_ + ntile * 16 + l15] = (_Float16)acc[r];
  }
}

// ---------------------------------------------------------------------------
// Kernel 4: flash attention.  8 waves/block = 8 consecutive 16-query tiles of
// one batch.  The block cooperatively double-buffers each 32-key K/V chunk in
// LDS via async loads; each wave then runs WMMA S = QK^T, online softmax, and
// WMMA P@V from LDS.
//   kbuf row stride 136 halves / vbuf row stride 40 halves: every 16-lane
//   group's ds_load_b128 covers all 64 LDS banks exactly once (conflict-free).
// ---------------------------------------------------------------------------
__global__ __launch_bounds__(256) void attn_kernel(const _Float16* __restrict__ qh,
                                                   const _Float16* __restrict__ kh,
                                                   const _Float16* __restrict__ vt,
                                                   float* __restrict__ out) {
  __shared__ _Float16 kbuf[2][32 * 136];   // K chunk:  [32 keys][128 h], padded
  __shared__ _Float16 vbuf[2][128 * 40];   // V^T chunk:[128 h][32 keys], padded
  __shared__ _Float16 pbuf[8 * 16 * 40];   // per-wave P re-layout tiles

  const int tid  = threadIdx.x;
  const int wave = tid >> 5;
  const int lane = tid & 31;
  const int g    = lane >> 4;
  const int l15  = lane & 15;
  const int task = blockIdx.x * 8 + wave;        // 0..1023
  const int b    = task >> 8;                    // batch (same for all 8 waves)
  const int q0   = (task & 255) * 16;            // this wave's first query row
  const int qb0  = ((blockIdx.x * 8) & 255) * 16;  // block's first query row
  const int nchunks = (qb0 + 128) >> 5;          // keys 0 .. qb0+127

  // log2(e) / sqrt(H): softmax in exp2 domain with the 1/sqrt(H) scale folded.
  const float kSc = 1.4426950408889634f * 0.08838834764831845f;

  const _Float16* kbase = kh + (size_t)b * T_ * H_;
  const _Float16* vbase = vt + (size_t)b * H_ * T_;

  // ---- load Q as four 16x32 A-tiles (K = H in chunks of 32), kept in regs --
  V16H aq[4];
  {
    const _Float16* qrow = qh + (size_t)(b * T_ + q0 + l15) * H_;
    #pragma unroll
    for (int c = 0; c < 4; ++c) {
      aq[c].u[0] = *(const uint4*)(qrow + c * 32 + 8 * g);
      aq[c].u[1] = *(const uint4*)(qrow + c * 32 + 8 * g + 16);
    }
  }

  v8f acc[8] = {};
  float m[8], l[8];
  #pragma unroll
  for (int r = 0; r < 8; ++r) { m[r] = -__builtin_inff(); l[r] = 0.0f; }

  // Cooperative async fill of one 32-key chunk: 512 x 16B for K, 512 for V.
  auto issue_chunk = [&](int slot, int key0) {
    #pragma unroll
    for (int it = 0; it < 2; ++it) {
      const int seg = tid + it * 256;                 // 0..511
      const int kr  = seg >> 4, kc8 = (seg & 15) * 8; // K: [32][128]
      cp16_to_lds(&kbuf[slot][kr * 136 + kc8],
                  kbase + (size_t)(key0 + kr) * H_ + kc8);
      const int vr  = seg >> 2, vc8 = (seg & 3) * 8;  // V^T: [128][32]
      cp16_to_lds(&vbuf[slot][vr * 40 + vc8],
                  vbase + (size_t)vr * T_ + key0 + vc8);
    }
  };

  issue_chunk(0, 0);
  _Float16* pb = pbuf + wave * (16 * 40);

  for (int c = 0; c < nchunks; ++c) {
    wait_lds_cp();       // my async writes for chunk c have landed in LDS
    __syncthreads();     // ...and everyone else's too
    if (c + 1 < nchunks) issue_chunk((c + 1) & 1, (c + 1) * 32);

    const int key0 = c * 32;
    if (key0 < q0 + 16) {               // inside this wave's causal range
      const _Float16* kc_ = kbuf[c & 1];
      const _Float16* vc_ = vbuf[c & 1];

      // ---- S = Q K^T for two 16-key subtiles, f32 accumulate --------------
      v8f s0 = {}, s1 = {};
      #pragma unroll
      for (int ch = 0; ch < 4; ++ch) {
        V16H bk0, bk1;
        const _Float16* kr0 = kc_ + (size_t)l15 * 136 + ch * 32 + 8 * g;
        const _Float16* kr1 = kr0 + 16 * 136;
        bk0.u[0] = *(const uint4*)kr0;   bk0.u[1] = *(const uint4*)(kr0 + 16);
        bk1.u[0] = *(const uint4*)kr1;   bk1.u[1] = *(const uint4*)(kr1 + 16);
        s0 = __builtin_amdgcn_wmma_f32_16x16x32_f16(false, aq[ch].v, false, bk0.v,
                                                    (short)0, s0, false, false);
        s1 = __builtin_amdgcn_wmma_f32_16x16x32_f16(false, aq[ch].v, false, bk1.v,
                                                    (short)0, s1, false, false);
      }

      // ---- causal mask + online softmax (rows in (r,g); cols in l15) ------
      const int col0 = key0 + l15;
      const int col1 = col0 + 16;
      const int rowg = q0 + 8 * g;
      float p0[8], p1[8], alpha[8];
      #pragma unroll
      for (int r = 0; r < 8; ++r) {
        const int rg = rowg + r;
        float v0 = (col0 <= rg) ? s0[r] * kSc : -__builtin_inff();
        float v1 = (col1 <= rg) ? s1[r] * kSc : -__builtin_inff();
        float mx = fmaxf(v0, v1);
        mx = fmaxf(mx, __shfl_xor(mx, 1, 32));
        mx = fmaxf(mx, __shfl_xor(mx, 2, 32));
        mx = fmaxf(mx, __shfl_xor(mx, 4, 32));
        mx = fmaxf(mx, __shfl_xor(mx, 8, 32));
        const float mn = fmaxf(m[r], mx);   // finite: each chunk has a valid
        alpha[r] = exp2f(m[r] - mn);        // column for every row (key0<=q0)
        m[r] = mn;
        p0[r] = exp2f(v0 - mn);
        p1[r] = exp2f(v1 - mn);
        float rs = p0[r] + p1[r];
        rs += __shfl_xor(rs, 1, 32);
        rs += __shfl_xor(rs, 2, 32);
        rs += __shfl_xor(rs, 4, 32);
        rs += __shfl_xor(rs, 8, 32);
        l[r] = l[r] * alpha[r] + rs;
      }

      // ---- re-layout P: C/D layout -> A layout via per-wave LDS tile ------
      #pragma unroll
      for (int r = 0; r < 8; ++r) {
        pb[(r + 8 * g) * 40 + l15]      = (_Float16)p0[r];
        pb[(r + 8 * g) * 40 + 16 + l15] = (_Float16)p1[r];
      }
      V16H ap;   // row M = l15; halves 0-7 -> cols 8g..8g+7, 8-15 -> +16
      ap.u[0] = *(const uint4*)(pb + l15 * 40 + 8 * g);
      ap.u[1] = *(const uint4*)(pb + l15 * 40 + 8 * g + 16);

      // ---- rescale O and accumulate P @ V ---------------------------------
      #pragma unroll
      for (int t = 0; t < 8; ++t)
        #pragma unroll
        for (int r = 0; r < 8; ++r) acc[t][r] *= alpha[r];

      #pragma unroll
      for (int t = 0; t < 8; ++t) {
        V16H bv;   // B col = h (16t + l15), K = key (contiguous in vbuf row)
        const _Float16* vr0 = vc_ + (size_t)(16 * t + l15) * 40 + 8 * g;
        bv.u[0] = *(const uint4*)vr0;
        bv.u[1] = *(const uint4*)(vr0 + 16);
        acc[t] = __builtin_amdgcn_wmma_f32_16x16x32_f16(false, ap.v, false, bv.v,
                                                        (short)0, acc[t], false, false);
      }
    }
  }

  // ---- epilogue: divide by row sums, store f32 [B,T,H] --------------------
  float inv[8];
  #pragma unroll
  for (int r = 0; r < 8; ++r) inv[r] = 1.0f / l[r];
  float* orow = out + (size_t)(b * T_ + q0 + 8 * g) * H_;
  #pragma unroll
  for (int t = 0; t < 8; ++t)
    #pragma unroll
    for (int r = 0; r < 8; ++r)
      orow[(size_t)r * H_ + 16 * t + l15] = acc[t][r] * inv[r];
}

// ---------------------------------------------------------------------------
// Host launcher.  Workspace layout (f16 elements):
//   xh [B*T*C] | wt [3*H*C] | kh [B*T*H] | qh [B*T*H] | vt [B*H*T]
//   total ~44.8 MB.
// ---------------------------------------------------------------------------
extern "C" void kernel_launch(void* const* d_in, const int* in_sizes, int n_in,
                              void* d_out, int out_size, void* d_ws, size_t ws_size,
                              hipStream_t stream) {
  const float* x  = (const float*)d_in[0];
  const float* wk = (const float*)d_in[1];
  const float* wq = (const float*)d_in[2];
  const float* wv = (const float*)d_in[3];
  float* out = (float*)d_out;

  _Float16* xh = (_Float16*)d_ws;
  _Float16* wt = xh + (size_t)B_ * T_ * C_;
  _Float16* kh = wt + (size_t)3 * H_ * C_;
  _Float16* qh = kh + (size_t)B_ * T_ * H_;
  _Float16* vt = qh + (size_t)B_ * T_ * H_;

  cvt_x_kernel<<<4096, 256, 0, stream>>>(x, xh, (B_ * T_ * C_) / 4);
  prep_w_kernel<<<512, 256, 0, stream>>>(wk, wq, wv, wt, 3 * H_ * C_);
  // 1024 M-tiles (8 per block) x 8 N-tiles x 3 matrices
  qkv_gemm_kernel<<<dim3(128, 8, 3), 256, 0, stream>>>(xh, wt, kh, qh, vt);
  // 1024 query-block tasks, 8 waves per block
  attn_kernel<<<128, 256, 0, stream>>>(qh, kh, vt, out);
}
